// AgentHetGNN_5119601016848
// MI455X (gfx1250) — matrix operand order repaired and checked by reading, not compiled, not run on hardware
//
#include <hip/hip_runtime.h>
#include <hip/hip_bf16.h>
#include <cmath>

// ---------------------------------------------------------------------------
// AgentHetGNN block for MI455X (gfx1250): all matmuls on v_wmma_f32_16x16x32_bf16
//  - GEMM: 64-wide N tile/block, 4 WMMAs per wave per K-step
//  - attention: flash-style, async global->LDS staging of V (ASYNCcnt path)
// ---------------------------------------------------------------------------

typedef __attribute__((ext_vector_type(16))) __bf16 v16bf;
typedef __attribute__((ext_vector_type(8)))  __bf16 v8bf;
typedef __attribute__((ext_vector_type(4)))  __bf16 v4bf;
typedef __attribute__((ext_vector_type(8)))  float  v8f;
typedef __attribute__((ext_vector_type(4)))  float  v4f;

#define WMMA_BF16 __builtin_amdgcn_wmma_f32_16x16x32_bf16

// Problem constants
#define BATCH 64
#define NA    128
#define NL    1024
#define NG    512
#define DMODEL 128
#define NHEAD 4
#define HEADD 32

__device__ __forceinline__ v16bf cat16(v8bf lo, v8bf hi) {
  v16bf r;
#pragma unroll
  for (int i = 0; i < 8; ++i) { r[i] = lo[i]; r[i + 8] = hi[i]; }
  return r;
}

// ---------------------------------------------------------------------------
// f32 -> bf16 flat convert
// ---------------------------------------------------------------------------
__global__ void f32_to_bf16(const float* __restrict__ x, __bf16* __restrict__ y, int n) {
  int i = blockIdx.x * 256 + threadIdx.x;
  if (i < n) y[i] = (__bf16)x[i];
}

// ---------------------------------------------------------------------------
// LayerNorm over rows of 128 f32, one wave32 per row, bf16 output
// ---------------------------------------------------------------------------
__global__ void __launch_bounds__(256)
layernorm_bf16(const float* __restrict__ X, const float* __restrict__ G,
               const float* __restrict__ Bt, __bf16* __restrict__ Y, int rows) {
  const int lane = threadIdx.x & 31;
  const int wave = threadIdx.x >> 5;
  const int row = blockIdx.x * 8 + wave;
  if (row >= rows) return;
  v4f x = *(const v4f*)(X + (size_t)row * DMODEL + lane * 4);
  float s  = x[0] + x[1] + x[2] + x[3];
  float sq = x[0]*x[0] + x[1]*x[1] + x[2]*x[2] + x[3]*x[3];
#pragma unroll
  for (int off = 1; off < 32; off <<= 1) {
    s  += __shfl_xor(s,  off, 32);
    sq += __shfl_xor(sq, off, 32);
  }
  float mean = s * (1.0f / DMODEL);
  float var  = sq * (1.0f / DMODEL) - mean * mean;
  float rst  = rsqrtf(var + 1e-5f);
  v4bf y;
#pragma unroll
  for (int j = 0; j < 4; ++j) {
    int c = lane * 4 + j;
    y[j] = (__bf16)((x[j] - mean) * rst * G[c] + Bt[c]);
  }
  *(v4bf*)(Y + (size_t)row * DMODEL + lane * 4) = y;
}

// ---------------------------------------------------------------------------
// WMMA GEMM: C[M,N] = act(A[M,K] @ W[K,N] + bias) (+ residual)
//  - block = 128 threads (4 waves); block tile = 64 rows x 64 cols
//  - wave computes 16x64: one A-fragment feeds 4 WMMAs per 32-K step
//  - W tile (32x64) staged transposed into LDS (stride 40 -> conflict-free b128)
// ---------------------------------------------------------------------------
template <int RELU, int OUT_BF16, int HAS_RES>
__global__ void __launch_bounds__(128)
gemm_bf16_wmma(const __bf16* __restrict__ A, int lda,
               const __bf16* __restrict__ W, int N,
               const float* __restrict__ bias,
               void* __restrict__ Cout, int ldc, int colOff,
               const float* __restrict__ Res,
               int M, int K) {
  __shared__ __align__(16) __bf16 wtile[64][40];
  const int lane  = threadIdx.x & 31;
  const int wave  = threadIdx.x >> 5;
  const int hi    = lane >> 4;     // half-wave select
  const int nl    = lane & 15;
  const int n0    = blockIdx.x * 64;
  const int m0    = (blockIdx.y * 4 + wave) * 16;

  // cooperative stage indices: thread covers [k][nn..nn+15] of the 32x64 tile
  const int sk = threadIdx.x >> 2;          // 0..31
  const int sn = (threadIdx.x & 3) * 16;    // 0,16,32,48

  v8f acc[4] = {v8f{}, v8f{}, v8f{}, v8f{}};
  for (int kb = 0; kb < K; kb += 32) {
    // ---- stage W[kb..kb+31][n0..n0+63] transposed into LDS ----------------
    {
      const __bf16* wp = W + (size_t)(kb + sk) * N + n0 + sn;
      v8bf w0 = *(const v8bf*)wp;
      v8bf w1 = *(const v8bf*)(wp + 8);
#pragma unroll
      for (int j = 0; j < 8; ++j) {
        wtile[sn + j][sk]     = w0[j];
        wtile[sn + 8 + j][sk] = w1[j];
      }
      if (kb + 32 < K)  // prefetch next W tile (global_prefetch_b8)
        __builtin_prefetch(W + (size_t)(kb + 32 + sk) * N + n0 + sn, 0, 0);
    }
    __syncthreads();
    // ---- A fragment: lane row = nl, K halves split 0..7 / 16..23 ----------
    const __bf16* apt = A + (size_t)(m0 + nl) * lda + kb + hi * 8;
    v16bf afrag = cat16(*(const v8bf*)apt, *(const v8bf*)(apt + 16));
    // ---- 4 column tiles: B frags from transposed LDS ----------------------
#pragma unroll
    for (int t = 0; t < 4; ++t) {
      const v8bf* bp = (const v8bf*)&wtile[t * 16 + nl][hi * 8];
      v16bf bfrag = cat16(bp[0], bp[2]);
      acc[t] = WMMA_BF16(false, afrag, false, bfrag, (short)0, acc[t], false, false);
    }
    __syncthreads();
  }
  // ---- epilogue (C layout: row = v + 8*hi, col = nl) ----------------------
#pragma unroll
  for (int t = 0; t < 4; ++t) {
    const int col = n0 + t * 16 + nl;
    const float bb = bias ? bias[col] : 0.0f;
#pragma unroll
    for (int v = 0; v < 8; ++v) {
      const size_t row = (size_t)(m0 + v + 8 * hi);
      float x = acc[t][v] + bb;
      if (RELU) x = fmaxf(x, 0.0f);
      if (HAS_RES) x += Res[row * ldc + colOff + col];
      if (OUT_BF16) ((__bf16*)Cout)[row * ldc + colOff + col] = (__bf16)x;
      else          ((float*) Cout)[row * ldc + colOff + col] = x;
    }
  }
}

// ---------------------------------------------------------------------------
// Flash-style attention, one wave per (batch, head, 16-query tile).
//   Q [B*NA, 384] bf16 (qcol selects relation slice), KV [B*Nk, 256] bf16
//   (K = cols 0..127, V = cols 128..255, head h at h*32).
//   d_k = 32 == WMMA K -> one WMMA per 16x16 score tile.
//   V chunk staged global->LDS with async copies (ASYNCcnt).
// ---------------------------------------------------------------------------
__global__ void __launch_bounds__(128)
attn_flash(const __bf16* __restrict__ Q, int qcol,
           const __bf16* __restrict__ KV,
           const unsigned char* __restrict__ mask,  // [B, NA, Nk] bool bytes
           __bf16* __restrict__ Out,                // [B*NA, 128]
           int Nk, float scale) {
  __shared__ __align__(16) __bf16 pstage[4][16][40];
  __shared__ __align__(16) __bf16 vstage[4][32][32];
  const int lane  = threadIdx.x & 31;
  const int wave  = threadIdx.x >> 5;
  const int hi    = lane >> 4;
  const int nl    = lane & 15;
  const int qtile = blockIdx.x * 4 + wave;
  const int h     = blockIdx.y;
  const int b     = blockIdx.z;
  const size_t qrow0 = (size_t)b * NA + qtile * 16;

  // Q fragment (A operand)
  const __bf16* qp = Q + (qrow0 + nl) * 384 + qcol + h * HEADD + hi * 8;
  const v16bf aq = cat16(*(const v8bf*)qp, *(const v8bf*)(qp + 16));

  const unsigned char* mrow = mask + ((size_t)b * NA + qtile * 16) * Nk;

  float Mx[8], Ls[8];
  v8f O0 = {}, O1 = {};
#pragma unroll
  for (int v = 0; v < 8; ++v) { Mx[v] = -3.0e38f; Ls[v] = 0.0f; }

  for (int kc = 0; kc < Nk; kc += 32) {
    // ---- kick off async stage of V chunk [32 keys][32 dims] into LDS ----
    {
      const __bf16* vrow = KV + (size_t)(b * Nk + kc + lane) * 256 + 128 + h * HEADD;
      unsigned ldsbase = (unsigned)(uintptr_t)&vstage[wave][lane][0];
#pragma unroll
      for (int j = 0; j < 4; ++j) {
        asm volatile("global_load_async_to_lds_b128 %0, %1, off"
                     :: "v"(ldsbase + j * 16), "v"(vrow + j * 8) : "memory");
      }
    }
    // ---- scores: two 16x16 tiles via two WMMAs -------------------------
    v8f s0 = {}, s1 = {};
    {
      const __bf16* kp = KV + (size_t)(b * Nk + kc + nl) * 256 + h * HEADD + hi * 8;
      v16bf bk = cat16(*(const v8bf*)kp, *(const v8bf*)(kp + 16));
      s0 = WMMA_BF16(false, aq, false, bk, (short)0, s0, false, false);
      const __bf16* kp2 = kp + (size_t)16 * 256;
      v16bf bk2 = cat16(*(const v8bf*)kp2, *(const v8bf*)(kp2 + 16));
      s1 = WMMA_BF16(false, aq, false, bk2, (short)0, s1, false, false);
    }
    // ---- online softmax update (row = v + 8*hi, col = nl / nl+16) ------
#pragma unroll
    for (int v = 0; v < 8; ++v) {
      const size_t mbase = (size_t)(v + 8 * hi) * Nk + kc;
      float x0 = s0[v] * scale;
      float x1 = s1[v] * scale;
      if (mrow[mbase + nl])       x0 = -1.0e10f;
      if (mrow[mbase + 16 + nl])  x1 = -1.0e10f;
      float mx = fmaxf(x0, x1);
#pragma unroll
      for (int off = 1; off < 16; off <<= 1) mx = fmaxf(mx, __shfl_xor(mx, off, 32));
      const float Mn = fmaxf(Mx[v], mx);
      const float alpha = __expf(Mx[v] - Mn);
      const float p0 = __expf(x0 - Mn);
      const float p1 = __expf(x1 - Mn);
      float rs = p0 + p1;
#pragma unroll
      for (int off = 1; off < 16; off <<= 1) rs += __shfl_xor(rs, off, 32);
      Ls[v] = Ls[v] * alpha + rs;
      O0[v] *= alpha;
      O1[v] *= alpha;
      Mx[v] = Mn;
      pstage[wave][v + 8 * hi][nl]      = (__bf16)p0;
      pstage[wave][v + 8 * hi][nl + 16] = (__bf16)p1;
    }
    asm volatile("s_wait_dscnt 0" ::: "memory");  // wave-local LDS RAW fence
    // ---- restage P from C-layout into A-layout -------------------------
    const __bf16* pr = &pstage[wave][nl][hi * 16];
    v16bf ap = cat16(*(const v8bf*)pr, *(const v8bf*)(pr + 8));
    // ---- V fragments from LDS (after async copies complete) ------------
    asm volatile("s_wait_asynccnt 0" ::: "memory");
    v16bf bv0, bv1;
#pragma unroll
    for (int i = 0; i < 16; ++i) {
      bv0[i] = vstage[wave][hi * 16 + i][nl];
      bv1[i] = vstage[wave][hi * 16 + i][nl + 16];
    }
    O0 = WMMA_BF16(false, ap, false, bv0, (short)0, O0, false, false);
    O1 = WMMA_BF16(false, ap, false, bv1, (short)0, O1, false, false);
  }
  // ---- finalize: divide by softmax denominator, write bf16 -------------
#pragma unroll
  for (int v = 0; v < 8; ++v) {
    const float inv = 1.0f / Ls[v];
    const size_t orow = (qrow0 + v + 8 * hi) * (size_t)DMODEL + h * HEADD;
    Out[orow + nl]      = (__bf16)(O0[v] * inv);
    Out[orow + nl + 16] = (__bf16)(O1[v] * inv);
  }
}

// ---------------------------------------------------------------------------
// SwiGLU gate: g = silu(u1) * u3, bf16 out
// ---------------------------------------------------------------------------
__global__ void silu_gate(const float* __restrict__ u1, const float* __restrict__ u3,
                          __bf16* __restrict__ g, int n) {
  int i = blockIdx.x * 256 + threadIdx.x;
  if (i < n) {
    float a = u1[i];
    g[i] = (__bf16)((a / (1.0f + __expf(-a))) * u3[i]);
  }
}

// ---------------------------------------------------------------------------
// Host orchestration
// ---------------------------------------------------------------------------
extern "C" void kernel_launch(void* const* d_in, const int* in_sizes, int n_in,
                              void* d_out, int out_size, void* d_ws, size_t ws_size,
                              hipStream_t stream) {
  (void)in_sizes; (void)n_in; (void)out_size; (void)ws_size;
  // inputs (setup_inputs dict order, params in insertion order)
  const float* agent = (const float*)d_in[0];
  const float* lanes = (const float*)d_in[1];
  const float* polys = (const float*)d_in[2];
  const unsigned char* a2a = (const unsigned char*)d_in[3];
  const unsigned char* l2a = (const unsigned char*)d_in[4];
  const unsigned char* g2a = (const unsigned char*)d_in[5];
  const float *lnq_g = (const float*)d_in[6],  *lnq_b = (const float*)d_in[7];
  const float *wq    = (const float*)d_in[8];
  const float *lno_g = (const float*)d_in[9],  *lno_b = (const float*)d_in[10];
  const float *wkvo  = (const float*)d_in[11];
  const float *lnl_g = (const float*)d_in[12], *lnl_b = (const float*)d_in[13];
  const float *wkvl  = (const float*)d_in[14];
  const float *lng_g = (const float*)d_in[15], *lng_b = (const float*)d_in[16];
  const float *wkvg  = (const float*)d_in[17];
  const float *fco_w = (const float*)d_in[18], *fco_b = (const float*)d_in[19];
  const float *fcl_w = (const float*)d_in[20], *fcl_b = (const float*)d_in[21];
  const float *fcg_w = (const float*)d_in[22], *fcg_b = (const float*)d_in[23];
  const float *fcs_w = (const float*)d_in[24], *fcs_b = (const float*)d_in[25];
  const float *wo_w  = (const float*)d_in[26], *wo_b  = (const float*)d_in[27];
  const float *fln_g = (const float*)d_in[28], *fln_b = (const float*)d_in[29];
  const float *w1_w  = (const float*)d_in[30], *w1_b  = (const float*)d_in[31];
  const float *w2_w  = (const float*)d_in[32], *w2_b  = (const float*)d_in[33];
  const float *w3_w  = (const float*)d_in[34], *w3_b  = (const float*)d_in[35];

  const int MA = BATCH * NA;   // 8192
  const int ML = BATCH * NL;   // 65536
  const int MG = BATCH * NG;   // 32768

  char* ws = (char*)d_ws;
  size_t off = 0;
  auto alloc = [&](size_t bytes) -> char* {
    char* p = ws + off;
    off = (off + bytes + 255) & ~(size_t)255;
    return p;
  };
  auto abf = [&](size_t elems) -> __bf16* { return (__bf16*)alloc(elems * 2); };
  auto af32 = [&](size_t elems) -> float* { return (float*)alloc(elems * 4); };

  __bf16* ln_q   = abf((size_t)MA * DMODEL);
  __bf16* ln_o   = abf((size_t)MA * DMODEL);
  __bf16* ln_l   = abf((size_t)ML * DMODEL);
  __bf16* ln_g   = abf((size_t)MG * DMODEL);
  __bf16* agentb = abf((size_t)MA * DMODEL);
  __bf16* qall   = abf((size_t)MA * 384);
  __bf16* kvo    = abf((size_t)MA * 256);
  __bf16* kvl    = abf((size_t)ML * 256);
  __bf16* kvg    = abf((size_t)MG * 256);
  __bf16* att_o  = abf((size_t)MA * DMODEL);
  __bf16* att_l  = abf((size_t)MA * DMODEL);
  __bf16* att_g  = abf((size_t)MA * DMODEL);
  __bf16* catb   = abf((size_t)MA * 512);
  __bf16* hn     = abf((size_t)MA * DMODEL);
  __bf16* gate   = abf((size_t)MA * 512);
  float*  hres   = af32((size_t)MA * DMODEL);
  float*  u1     = af32((size_t)MA * 512);
  float*  u3     = af32((size_t)MA * 512);
  // bf16 weights
  __bf16* bwq   = abf(128 * 384);
  __bf16* bwkvo = abf(128 * 256);
  __bf16* bwkvl = abf(128 * 256);
  __bf16* bwkvg = abf(128 * 256);
  __bf16* bfco  = abf(128 * 128);
  __bf16* bfcl  = abf(128 * 128);
  __bf16* bfcg  = abf(128 * 128);
  __bf16* bfcs  = abf(128 * 128);
  __bf16* bwo   = abf(512 * 128);
  __bf16* bw1   = abf(128 * 512);
  __bf16* bw2   = abf(512 * 128);
  __bf16* bw3   = abf(128 * 512);

  auto cvt = [&](const float* src, __bf16* dst, int n) {
    f32_to_bf16<<<(n + 255) / 256, 256, 0, stream>>>(src, dst, n);
  };
  cvt(wq, bwq, 128 * 384);
  cvt(wkvo, bwkvo, 128 * 256);
  cvt(wkvl, bwkvl, 128 * 256);
  cvt(wkvg, bwkvg, 128 * 256);
  cvt(fco_w, bfco, 128 * 128);
  cvt(fcl_w, bfcl, 128 * 128);
  cvt(fcg_w, bfcg, 128 * 128);
  cvt(fcs_w, bfcs, 128 * 128);
  cvt(wo_w, bwo, 512 * 128);
  cvt(w1_w, bw1, 128 * 512);
  cvt(w2_w, bw2, 512 * 128);
  cvt(w3_w, bw3, 128 * 512);
  cvt(agent, agentb, MA * DMODEL);

  // layernorms
  layernorm_bf16<<<MA / 8, 256, 0, stream>>>(agent, lnq_g, lnq_b, ln_q, MA);
  layernorm_bf16<<<MA / 8, 256, 0, stream>>>(agent, lno_g, lno_b, ln_o, MA);
  layernorm_bf16<<<ML / 8, 256, 0, stream>>>(lanes, lnl_g, lnl_b, ln_l, ML);
  layernorm_bf16<<<MG / 8, 256, 0, stream>>>(polys, lng_g, lng_b, ln_g, MG);

  // projections: q [*,384]; kv [*,256]
  gemm_bf16_wmma<0, 1, 0><<<dim3(384 / 64, MA / 64), 128, 0, stream>>>(
      ln_q, 128, bwq, 384, nullptr, qall, 384, 0, nullptr, MA, 128);
  gemm_bf16_wmma<0, 1, 0><<<dim3(256 / 64, MA / 64), 128, 0, stream>>>(
      ln_o, 128, bwkvo, 256, nullptr, kvo, 256, 0, nullptr, MA, 128);
  gemm_bf16_wmma<0, 1, 0><<<dim3(256 / 64, ML / 64), 128, 0, stream>>>(
      ln_l, 128, bwkvl, 256, nullptr, kvl, 256, 0, nullptr, ML, 128);
  gemm_bf16_wmma<0, 1, 0><<<dim3(256 / 64, MG / 64), 128, 0, stream>>>(
      ln_g, 128, bwkvg, 256, nullptr, kvg, 256, 0, nullptr, MG, 128);

  // attention (scale = log_32(Nk+1) / sqrt(d_k))
  const float s_o = logf((float)NA + 1.0f) / logf(32.0f) / sqrtf((float)HEADD);
  const float s_l = logf((float)NL + 1.0f) / logf(32.0f) / sqrtf((float)HEADD);
  const float s_g = logf((float)NG + 1.0f) / logf(32.0f) / sqrtf((float)HEADD);
  dim3 agrid(NA / 16 / 4, NHEAD, BATCH);
  attn_flash<<<agrid, 128, 0, stream>>>(qall, 0,   kvo, a2a, att_o, NA, s_o);
  attn_flash<<<agrid, 128, 0, stream>>>(qall, 128, kvl, l2a, att_l, NL, s_l);
  attn_flash<<<agrid, 128, 0, stream>>>(qall, 256, kvg, g2a, att_g, NG, s_g);

  // fused FC+ReLU straight into concat buffer [self|other|l2a|g2a]
  dim3 fgrid(128 / 64, MA / 64);
  gemm_bf16_wmma<1, 1, 0><<<fgrid, 128, 0, stream>>>(
      agentb, 128, bfcs, 128, fcs_b, catb, 512, 0, nullptr, MA, 128);
  gemm_bf16_wmma<1, 1, 0><<<fgrid, 128, 0, stream>>>(
      att_o, 128, bfco, 128, fco_b, catb, 512, 128, nullptr, MA, 128);
  gemm_bf16_wmma<1, 1, 0><<<fgrid, 128, 0, stream>>>(
      att_l, 128, bfcl, 128, fcl_b, catb, 512, 256, nullptr, MA, 128);
  gemm_bf16_wmma<1, 1, 0><<<fgrid, 128, 0, stream>>>(
      att_g, 128, bfcg, 128, fcg_b, catb, 512, 384, nullptr, MA, 128);

  // out_fc -> h (f32, kept for residual)
  gemm_bf16_wmma<0, 0, 0><<<dim3(128 / 64, MA / 64), 128, 0, stream>>>(
      catb, 512, bwo, 128, wo_b, hres, 128, 0, nullptr, MA, 512);

  // FFN: ln, two 128->512 GEMMs, SwiGLU gate, 512->128 GEMM + residual
  layernorm_bf16<<<MA / 8, 256, 0, stream>>>(hres, fln_g, fln_b, hn, MA);
  gemm_bf16_wmma<0, 0, 0><<<dim3(512 / 64, MA / 64), 128, 0, stream>>>(
      hn, 128, bw1, 512, w1_b, u1, 512, 0, nullptr, MA, 128);
  gemm_bf16_wmma<0, 0, 0><<<dim3(512 / 64, MA / 64), 128, 0, stream>>>(
      hn, 128, bw3, 512, w3_b, u3, 512, 0, nullptr, MA, 128);
  silu_gate<<<(MA * 512 + 255) / 256, 256, 0, stream>>>(u1, u3, gate, MA * 512);
  gemm_bf16_wmma<0, 0, 1><<<dim3(128 / 64, MA / 64), 128, 0, stream>>>(
      gate, 512, bw2, 128, w2_b, (float*)d_out, 128, 0, hres, MA, 512);
}
